// EEGCNNTransformer_52707838656784
// MI455X (gfx1250) — matrix operand order repaired and checked by reading, not compiled, not run on hardware
//
#include <hip/hip_runtime.h>
#include <hip/hip_bf16.h>
#include <math.h>

// ---- problem constants (match reference) ----
#define B_   64
#define C_   122
#define T_   500
#define ED_  64
#define KS_  25
#define L_   4
#define H_   8
#define DFF_ 256
#define HH_  128
#define OD_  20
#define NS_  13
#define S_   123         // C_ + 1 (cls token)
#define EPS_ 1e-5f
#define RL_  560         // padded act1 row length: 12 halo + 512 + 31 tail max -> 560 ok
#define K2_  1024        // conv2 GEMM K, padded: k = c*32 + j, j in [0,32), taps >=25 are zero

typedef __attribute__((ext_vector_type(16))) _Float16 v16h_t;
typedef __attribute__((ext_vector_type(8)))  _Float16 v8h_t;
typedef __attribute__((ext_vector_type(8)))  float    v8f_t;

__device__ __forceinline__ float geluf(float x) {
    return 0.5f * x * (1.0f + erff(x * 0.70710678118654752f));
}
__device__ __forceinline__ v16h_t combine16(v8h_t a, v8h_t b) {
    return __builtin_shufflevector(a, b, 0,1,2,3,4,5,6,7,8,9,10,11,12,13,14,15);
}

// ---------------- f32 -> f16 conversion ----------------
__global__ void cvt_f32_to_f16_kernel(const float* __restrict__ in, _Float16* __restrict__ out, int n) {
    int i = blockIdx.x * blockDim.x + threadIdx.x;
    if (i < n) out[i] = (_Float16)in[i];
}

// ---------------- conv2 weights f32[64][32][25] -> f16[64][1024], k = c*32+j, zero-padded ----------------
__global__ void pad_conv2_w_kernel(const float* __restrict__ w, _Float16* __restrict__ wp) {
    int i = blockIdx.x * blockDim.x + threadIdx.x;
    if (i < 64 * K2_) {
        int m = i >> 10, r = i & (K2_ - 1), c = r >> 5, j = r & 31;
        wp[i] = (j < KS_) ? (_Float16)w[(m * 32 + c) * KS_ + j] : (_Float16)0.0f;
    }
}

// ---------------- cls token init ----------------
__global__ void cls_init_kernel(const float* __restrict__ cls, float* __restrict__ x) {
    int i = blockIdx.x * blockDim.x + threadIdx.x;
    if (i < B_ * ED_) {
        int b = i / ED_, d = i % ED_;
        x[(size_t)b * S_ * ED_ + d] = cls[d];
    }
}

// ---------------- fused conv1+BN+GELU -> conv2(WMMA)+BN+GELU -> mean-pool + pos ----------------
// one block per (b,c) row of the EEG input
__global__ __launch_bounds__(256) void conv_stage_kernel(
    const float* __restrict__ eeg,
    const float* __restrict__ w1c, const float* __restrict__ b1c,
    const float* __restrict__ g1,  const float* __restrict__ bb1,
    const float* __restrict__ m1,  const float* __restrict__ v1,
    const _Float16* __restrict__ W2p,            // [64][1024] f16, k = c*32+j zero-padded
    const float* __restrict__ b2c,
    const float* __restrict__ g2,  const float* __restrict__ bb2,
    const float* __restrict__ m2,  const float* __restrict__ v2,
    const float* __restrict__ pos,               // [C][ED]
    float* __restrict__ x)                       // [B][S][ED]
{
    __shared__ float    eegpad[T_ + 28];         // index p = t + 12, zero halo
    __shared__ float    w1s[32 * KS_];
    __shared__ _Float16 act1[32 * RL_];          // f16 conv1 activations, zero-padded
    __shared__ float    pooled[ED_];

    const int tid = threadIdx.x;
    const int samp = blockIdx.x;                 // 0 .. B*C-1
    const int b = samp / C_;
    const int c = samp % C_;

    // zero-fill LDS
    for (int i = tid; i < 32 * RL_; i += 256) act1[i] = (_Float16)0.0f;
    for (int i = tid; i < T_ + 28;  i += 256) eegpad[i] = 0.0f;
    if (tid < ED_) pooled[tid] = 0.0f;
    __syncthreads();

    // stage eeg row + conv1 weights
    for (int i = tid; i < T_; i += 256) eegpad[12 + i] = eeg[(size_t)samp * T_ + i];
    for (int i = tid; i < 32 * KS_; i += 256) w1s[i] = w1c[i];
    __syncthreads();

    // conv1 + bn1 + gelu -> act1 (f16)
    for (int idx = tid; idx < 32 * T_; idx += 256) {
        int oc = idx / T_;
        int t  = idx - oc * T_;
        const float* wr = &w1s[oc * KS_];
        float s = 0.0f;
        #pragma unroll
        for (int j = 0; j < KS_; ++j) s += wr[j] * eegpad[t + j];
        float sg  = g1[oc] * rsqrtf(v1[oc] + EPS_);
        float val = s * sg + (b1c[oc] - m1[oc]) * sg + bb1[oc];
        act1[oc * RL_ + 12 + t] = (_Float16)geluf(val);
    }
    __syncthreads();

    // conv2 as WMMA GEMM: out[64][512] = W2p[64][1024] x im2col[1024][512]
    // K-layout k = c*32 + j, so each 32-wide K chunk is one channel and each lane's
    // 16 B halves are contiguous in the act1 row -> pure immediate-offset LDS loads.
    const int w    = tid >> 5;
    const int lane = tid & 31;
    const int lo   = lane & 15;
    const int hi   = (lane >> 4) & 1;
    const int m0    = (w & 3) << 4;              // M tile (out channel base)
    const int nbase = (w >> 2) << 4;             // 16 N-tiles per wave group

    float bnA[8], bnB[8], sums[8];
    #pragma unroll
    for (int r = 0; r < 8; ++r) {
        int ch = m0 + r + hi * 8;
        float sg = g2[ch] * rsqrtf(v2[ch] + EPS_);
        bnA[r] = sg;
        bnB[r] = (b2c[ch] - m2[ch]) * sg + bb2[ch];
        sums[r] = 0.0f;
    }

    const _Float16* arow = W2p + (size_t)(m0 + lo) * K2_;
    const int hi8  = hi * 8;
    const int hi16 = hi * 16;

    for (int g = 0; g < 4; ++g) {                // groups of 4 N-tiles share A fragments
        const int tile0 = nbase + g * 4;
        v8f_t accs[4] = {{}, {}, {}, {}};
        for (int kc = 0; kc < K2_; kc += 32) {
            __builtin_prefetch(arow + kc + 64, 0, 0);
            v8h_t a0 = *(const v8h_t*)(arow + kc + hi16);
            v8h_t a1 = *(const v8h_t*)(arow + kc + hi16 + 8);
            v16h_t a = combine16(a0, a1);
            const _Float16* bbase = &act1[((kc >> 5) * RL_) + hi16 + lo];
            #pragma unroll
            for (int t = 0; t < 4; ++t) {
                const _Float16* bptr = bbase + ((tile0 + t) << 4);
                v16h_t bm;
                #pragma unroll
                for (int i = 0; i < 16; ++i) bm[i] = bptr[i];
                accs[t] = __builtin_amdgcn_wmma_f32_16x16x32_f16(false, a, false, bm,
                                                                 (short)0, accs[t], false, false);
            }
        }
        #pragma unroll
        for (int t = 0; t < 4; ++t) {
            const int n = ((tile0 + t) << 4) + lo;
            if (n < T_) {
                #pragma unroll
                for (int r = 0; r < 8; ++r)
                    sums[r] += geluf(accs[t][r] * bnA[r] + bnB[r]);
            }
        }
    }
    #pragma unroll
    for (int r = 0; r < 8; ++r)
        atomicAdd(&pooled[m0 + r + hi8], sums[r]);
    __syncthreads();

    if (tid < ED_) {
        float val = pooled[tid] * (1.0f / (float)T_) + pos[c * ED_ + tid];
        x[((size_t)b * S_ + 1 + c) * ED_ + tid] = val;
    }
}

// ---------------- generic WMMA GEMM: out = A[M,K] * Bw[N,K]^T (+bias)(gelu?)(+resid) ----------------
template<int GELU_EPI>
__global__ __launch_bounds__(256) void gemm_wmma_kernel(
    const _Float16* __restrict__ A, int lda,
    const _Float16* __restrict__ Bw, int ldb,
    const float* __restrict__ bias,
    const float* __restrict__ resid,
    float* __restrict__ outF,
    _Float16* __restrict__ outH,
    int M, int N, int K, int ldo)
{
    const int lane = threadIdx.x & 31;
    const int lo   = lane & 15;
    const int hi   = (lane >> 4) & 1;
    const int tw   = blockIdx.x * (blockDim.x >> 5) + (threadIdx.x >> 5);
    const int tilesN = N >> 4;
    const int tilesM = M >> 4;
    if (tw >= tilesM * tilesN) return;            // wave-uniform exit (EXEC stays all-ones)
    const int m0 = (tw / tilesN) << 4;
    const int n0 = (tw % tilesN) << 4;

    v8f_t acc = {};
    const _Float16* arow = A  + (size_t)(m0 + lo) * lda;
    const _Float16* brow = Bw + (size_t)(n0 + lo) * ldb + hi * 16;
    for (int kc = 0; kc < K; kc += 32) {
        v8h_t a0 = *(const v8h_t*)(arow + kc + hi * 16);
        v8h_t a1 = *(const v8h_t*)(arow + kc + hi * 16 + 8);
        v8h_t b0 = *(const v8h_t*)(brow + kc);
        v8h_t b1 = *(const v8h_t*)(brow + kc + 8);
        v16h_t a = combine16(a0, a1);
        v16h_t bm = combine16(b0, b1);
        acc = __builtin_amdgcn_wmma_f32_16x16x32_f16(false, a, false, bm,
                                                     (short)0, acc, false, false);
    }
    const int n  = n0 + lo;
    const float bn = bias ? bias[n] : 0.0f;
    #pragma unroll
    for (int r = 0; r < 8; ++r) {
        int m = m0 + r + hi * 8;
        float v = acc[r] + bn;
        if (GELU_EPI) v = geluf(v);
        size_t idx = (size_t)m * ldo + n;
        if (resid) v += resid[idx];
        if (outF) outF[idx] = v;
        if (outH) outH[idx] = (_Float16)v;
    }
}

// ---------------- layernorm over last dim (64) ----------------
__global__ void ln_kernel(const float* __restrict__ x,
                          const float* __restrict__ g, const float* __restrict__ bb,
                          _Float16* __restrict__ outH, float* __restrict__ outF, int rows)
{
    int r = blockIdx.x * blockDim.x + threadIdx.x;
    if (r >= rows) return;
    const float* row = x + (size_t)r * ED_;
    float m = 0.0f;
    for (int d = 0; d < ED_; ++d) m += row[d];
    m *= (1.0f / ED_);
    float v = 0.0f;
    for (int d = 0; d < ED_; ++d) { float t = row[d] - m; v += t * t; }
    v *= (1.0f / ED_);
    float s = rsqrtf(v + EPS_);
    for (int d = 0; d < ED_; ++d) {
        float y = (row[d] - m) * s * g[d] + bb[d];
        if (outH) outH[(size_t)r * ED_ + d] = (_Float16)y;
        if (outF) outF[(size_t)r * ED_ + d] = y;
    }
}

// ---------------- attention: one block per (b,h), online softmax, dh=8 ----------------
__global__ __launch_bounds__(128) void attn_kernel(const float* __restrict__ qkv,
                                                   _Float16* __restrict__ oh)
{
    const int bh = blockIdx.x;
    const int b = bh / H_, h = bh % H_;
    __shared__ float ks[S_ * 8];
    __shared__ float vs[S_ * 8];
    const float* base = qkv + (size_t)b * S_ * (3 * ED_);
    for (int i = threadIdx.x; i < S_ * 8; i += 128) {
        int s = i >> 3, d = i & 7;
        ks[i] = base[s * (3 * ED_) + ED_     + h * 8 + d];
        vs[i] = base[s * (3 * ED_) + 2 * ED_ + h * 8 + d];
    }
    __syncthreads();
    const int i = threadIdx.x;
    if (i < S_) {
        float q[8];
        #pragma unroll
        for (int d = 0; d < 8; ++d)
            q[d] = base[i * (3 * ED_) + h * 8 + d] * 0.35355339059327373f; // 1/sqrt(8)
        float mx = -1e30f, sum = 0.0f, o[8];
        #pragma unroll
        for (int d = 0; d < 8; ++d) o[d] = 0.0f;
        for (int kk = 0; kk < S_; ++kk) {
            float dot = 0.0f;
            #pragma unroll
            for (int d = 0; d < 8; ++d) dot += q[d] * ks[kk * 8 + d];
            float nm = fmaxf(mx, dot);
            float corr = expf(mx - nm);
            float w = expf(dot - nm);
            sum = sum * corr + w;
            #pragma unroll
            for (int d = 0; d < 8; ++d) o[d] = o[d] * corr + w * vs[kk * 8 + d];
            mx = nm;
        }
        float inv = 1.0f / sum;
        #pragma unroll
        for (int d = 0; d < 8; ++d)
            oh[((size_t)b * S_ + i) * ED_ + h * 8 + d] = (_Float16)(o[d] * inv);
    }
}

// ---------------- subject-indexed head MLP ----------------
__global__ __launch_bounds__(128) void head_kernel(
    const float* __restrict__ xlnf, const int* __restrict__ sidx,
    const float* __restrict__ w1, const float* __restrict__ b1,
    const float* __restrict__ w2, const float* __restrict__ b2,
    float* __restrict__ out)
{
    const int b = blockIdx.x;
    const int s = sidx[b];
    __shared__ float cls[ED_];
    __shared__ float hb[HH_];
    if (threadIdx.x < ED_) cls[threadIdx.x] = xlnf[(size_t)b * S_ * ED_ + threadIdx.x];
    __syncthreads();
    const int j = threadIdx.x;
    {
        const float* wr = w1 + ((size_t)s * HH_ + j) * ED_;
        float acc = b1[s * HH_ + j];
        for (int d = 0; d < ED_; ++d) acc += wr[d] * cls[d];
        hb[j] = geluf(acc);
    }
    __syncthreads();
    if (j < OD_) {
        const float* wr = w2 + ((size_t)s * OD_ + j) * HH_;
        float acc = b2[s * OD_ + j];
        for (int t = 0; t < HH_; ++t) acc += wr[t] * hb[t];
        out[b * OD_ + j] = acc;
    }
}

// =====================================================================
extern "C" void kernel_launch(void* const* d_in, const int* in_sizes, int n_in,
                              void* d_out, int out_size, void* d_ws, size_t ws_size,
                              hipStream_t stream) {
    const float* eeg     = (const float*)d_in[0];
    const int*   sidx    = (const int*)  d_in[1];
    const float* conv1_w = (const float*)d_in[2];
    const float* conv1_b = (const float*)d_in[3];
    const float* bn1_g   = (const float*)d_in[4];
    const float* bn1_b   = (const float*)d_in[5];
    const float* bn1_m   = (const float*)d_in[6];
    const float* bn1_v   = (const float*)d_in[7];
    const float* conv2_w = (const float*)d_in[8];
    const float* conv2_b = (const float*)d_in[9];
    const float* bn2_g   = (const float*)d_in[10];
    const float* bn2_b   = (const float*)d_in[11];
    const float* bn2_m   = (const float*)d_in[12];
    const float* bn2_v   = (const float*)d_in[13];
    const float* pos     = (const float*)d_in[14];
    const float* clstok  = (const float*)d_in[15];
    const float* wqkv    = (const float*)d_in[16];
    const float* bqkv    = (const float*)d_in[17];
    const float* wo      = (const float*)d_in[18];
    const float* bo      = (const float*)d_in[19];
    const float* ln1_g   = (const float*)d_in[20];
    const float* ln1_b   = (const float*)d_in[21];
    const float* ln2_g   = (const float*)d_in[22];
    const float* ln2_b   = (const float*)d_in[23];
    const float* ffn_w1  = (const float*)d_in[24];
    const float* ffn_b1  = (const float*)d_in[25];
    const float* ffn_w2  = (const float*)d_in[26];
    const float* ffn_b2  = (const float*)d_in[27];
    const float* lnf_g   = (const float*)d_in[28];
    const float* lnf_b   = (const float*)d_in[29];
    const float* head_w1 = (const float*)d_in[30];
    const float* head_b1 = (const float*)d_in[31];
    const float* head_w2 = (const float*)d_in[32];
    const float* head_b2 = (const float*)d_in[33];
    float* out = (float*)d_out;

    // workspace bump allocator (256B aligned)
    char* wp = (char*)d_ws;
    auto alloc = [&](size_t bytes) -> void* {
        void* r = (void*)wp;
        wp += (bytes + 255) & ~((size_t)255);
        return r;
    };
    const int TOK = B_ * S_;                                   // 7872 tokens
    _Float16* W2p   = (_Float16*)alloc(64 * K2_ * 2);          // conv2 weights f16 (padded K)
    _Float16* wqkvh = (_Float16*)alloc(L_ * 3*ED_*ED_ * 2);
    _Float16* woh   = (_Float16*)alloc(L_ * ED_*ED_ * 2);
    _Float16* w1h   = (_Float16*)alloc(L_ * DFF_*ED_ * 2);
    _Float16* w2h   = (_Float16*)alloc(L_ * ED_*DFF_ * 2);
    float*    x     = (float*)   alloc((size_t)TOK * ED_ * 4);
    _Float16* yh    = (_Float16*)alloc((size_t)TOK * ED_ * 2);
    float*    qkvb  = (float*)   alloc((size_t)TOK * 3*ED_ * 4);
    _Float16* ohb   = (_Float16*)alloc((size_t)TOK * ED_ * 2);
    _Float16* ffh   = (_Float16*)alloc((size_t)TOK * DFF_ * 2);
    float*    xlnf  = (float*)   alloc((size_t)TOK * ED_ * 4);

    auto cvt = [&](const float* in, _Float16* o, int n) {
        cvt_f32_to_f16_kernel<<<(n + 255) / 256, 256, 0, stream>>>(in, o, n);
    };
    pad_conv2_w_kernel<<<(64 * K2_ + 255) / 256, 256, 0, stream>>>(conv2_w, W2p);
    cvt(wqkv,   wqkvh, L_ * 3*ED_*ED_);
    cvt(wo,     woh,   L_ * ED_*ED_);
    cvt(ffn_w1, w1h,   L_ * DFF_*ED_);
    cvt(ffn_w2, w2h,   L_ * ED_*DFF_);

    cls_init_kernel<<<(B_ * ED_ + 255) / 256, 256, 0, stream>>>(clstok, x);

    // fused conv stack (dominant cost, WMMA)
    conv_stage_kernel<<<B_ * C_, 256, 0, stream>>>(
        eeg, conv1_w, conv1_b, bn1_g, bn1_b, bn1_m, bn1_v,
        W2p, conv2_b, bn2_g, bn2_b, bn2_m, bn2_v, pos, x);

    auto gemm = [&](int epi, const _Float16* A, int lda, const _Float16* Bw, int ldb,
                    const float* bias, const float* resid, float* oF, _Float16* oH,
                    int M, int N, int K) {
        int tiles = (M >> 4) * (N >> 4);
        int blocks = (tiles + 7) / 8;
        if (epi)
            gemm_wmma_kernel<1><<<blocks, 256, 0, stream>>>(A, lda, Bw, ldb, bias, resid, oF, oH, M, N, K, N);
        else
            gemm_wmma_kernel<0><<<blocks, 256, 0, stream>>>(A, lda, Bw, ldb, bias, resid, oF, oH, M, N, K, N);
    };

    for (int l = 0; l < L_; ++l) {
        // y = LN1(x) -> f16
        ln_kernel<<<(TOK + 255) / 256, 256, 0, stream>>>(x, ln1_g + l*ED_, ln1_b + l*ED_, yh, nullptr, TOK);
        // qkv = y @ wqkv^T + bqkv
        gemm(0, yh, ED_, wqkvh + (size_t)l*3*ED_*ED_, ED_, bqkv + l*3*ED_, nullptr, qkvb, nullptr,
             TOK, 3*ED_, ED_);
        // attention -> o (f16)
        attn_kernel<<<B_ * H_, 128, 0, stream>>>(qkvb, ohb);
        // x = x + o @ wo^T + bo
        gemm(0, ohb, ED_, woh + (size_t)l*ED_*ED_, ED_, bo + l*ED_, x, x, nullptr,
             TOK, ED_, ED_);
        // y = LN2(x) -> f16
        ln_kernel<<<(TOK + 255) / 256, 256, 0, stream>>>(x, ln2_g + l*ED_, ln2_b + l*ED_, yh, nullptr, TOK);
        // ff = gelu(y @ w1^T + b1) -> f16
        gemm(1, yh, ED_, w1h + (size_t)l*DFF_*ED_, ED_, ffn_b1 + l*DFF_, nullptr, nullptr, ffh,
             TOK, DFF_, ED_);
        // x = x + ff @ w2^T + b2
        gemm(0, ffh, DFF_, w2h + (size_t)l*ED_*DFF_, DFF_, ffn_b2 + l*ED_, x, x, nullptr,
             TOK, ED_, DFF_);
    }

    // final LN (f32) and subject head
    ln_kernel<<<(TOK + 255) / 256, 256, 0, stream>>>(x, lnf_g, lnf_b, nullptr, xlnf, TOK);
    head_kernel<<<B_, 128, 0, stream>>>(xlnf, sidx, head_w1, head_b1, head_w2, head_b2, out);

    (void)in_sizes; (void)n_in; (void)out_size; (void)ws_size;
}
